// BidirectionAttention_17085379904046
// MI455X (gfx1250) — matrix operand order repaired and checked by reading, not compiled, not run on hardware
//
#include <hip/hip_runtime.h>
#include <cstddef>
#include <cstdint>

typedef __attribute__((ext_vector_type(16))) _Float16 v16h;
typedef __attribute__((ext_vector_type(8)))  _Float16 v8h;
typedef __attribute__((ext_vector_type(8)))  float    v8f;
typedef __attribute__((ext_vector_type(4)))  float    v4f;

#define HEADS  4
#define DHEAD  64
#define NFEAT  9216   /* 96*96 */
#define NMAP   256    /* 16*16 */
#define CIN    256
#define NBATCH 4
#define NBH    16     /* NBATCH * HEADS */
#define ATTN_SCALE 0.125f

// ---------------------------------------------------------------------------
// WMMA helpers: D = A(16x32 f16) * B(32x16 f16) + C(16x16 f32), wave32.
// Tiles staged in LDS in *fragment-ready* order: lane L's v16h fragment is the
// 32 contiguous bytes at tile[L*16] (one aligned v16h load = 2x ds_load_b128).
// Per CDNA5 ISA 7.12.2:
//   A (m,k): lane = m + 16*k[3], slot i = (k&7) + 8*(k>=16)
//   B (k,n): lane = n + 16*k[3], slot i = (k&7) + 8*(k>=16)
// Aligned 8-long k-runs map to 8 *contiguous* slots in one lane-row for both
// layouts, so staging along K uses single 16B ds_store_b128 stores.
// ---------------------------------------------------------------------------
__device__ __forceinline__ v8f wmma_f16(v16h a, v16h b, v8f c) {
  return __builtin_amdgcn_wmma_f32_16x16x32_f16(false, a, false, b, (short)0, c,
                                                false, false);
}

__device__ __forceinline__ int frag_slot_a(int m, int k) {
  int half = (k >> 3) & 1;
  int i = (k & 7) | ((k & 16) >> 1);  // +8 when k >= 16
  return (m + 16 * half) * 16 + i;
}

__device__ __forceinline__ int frag_slot_b(int k, int n) {
  return (n + 16 * ((k >> 3) & 1)) * 16 + ((k & 7) | ((k & 16) >> 1));
}

__device__ __forceinline__ v16h lds_frag(const _Float16* p) {
  return *(const v16h*)p;
}

__device__ __forceinline__ v8h cvt_v8h(v4f a, v4f b) {
  v8h r;
  r[0] = (_Float16)a[0]; r[1] = (_Float16)a[1];
  r[2] = (_Float16)a[2]; r[3] = (_Float16)a[3];
  r[4] = (_Float16)b[0]; r[5] = (_Float16)b[1];
  r[6] = (_Float16)b[2]; r[7] = (_Float16)b[3];
  return r;
}

// ---------------------------------------------------------------------------
// Kernel 1: QV projection (1x1 conv). C[512, Ntok] = Wqv[512,256] * X[b][256][Ntok]
// Outputs split to q/v in head-split f16 layout [b*4+h][Ntok][64], c = d*4 + h.
// Block: 256 thr, tile 128 oc x 64 tok, K=256 in steps of 32.
// ---------------------------------------------------------------------------
__global__ void qv_proj_kernel(const float* __restrict__ X,
                               const float* __restrict__ Wqv,
                               _Float16* __restrict__ q_out,
                               _Float16* __restrict__ v_out,
                               int Ntok) {
  __shared__ alignas(32) _Float16 sA[8 * 512];  // 8 A-tiles (oc-subtile)
  __shared__ alignas(32) _Float16 sB[4 * 512];  // 4 B-tiles (tok-subtile)
  const int tid = threadIdx.x;
  const int w = tid >> 5, lane = tid & 31;
  const int half = lane >> 4, ln = lane & 15;
  const int px0 = blockIdx.x * 64;
  const int oc0 = blockIdx.y * 128;
  const int b = blockIdx.z;

  v8f acc[4] = {};
  for (int k0 = 0; k0 < CIN; k0 += 32) {
    const float* WqvB = Wqv + oc0 * CIN + k0;
    const float* Xb = X + ((size_t)b * CIN + k0) * Ntok + px0;
    __syncthreads();
#pragma unroll
    for (int i = 0; i < 2; ++i) {  // A: 128 oc x 32 k, 8-wide runs
      int r = tid + i * 256;
      int oc = r >> 2, k8 = (r & 3) * 8;
      const float* src = WqvB + oc * CIN + k8;
      *(v8h*)&sA[(oc >> 4) * 512 + frag_slot_a(oc & 15, k8)] =
          cvt_v8h(*(const v4f*)src, *(const v4f*)(src + 4));
    }
#pragma unroll
    for (int i = 0; i < 2; ++i) {  // B: 32 k x 64 tok, float4 along tok
      int r = tid + i * 256;
      int kk = r >> 4, px4 = (r & 15) * 4;
      v4f x = *(const v4f*)(Xb + kk * Ntok + px4);
#pragma unroll
      for (int e = 0; e < 4; ++e) {
        int px = px4 + e;
        sB[(px >> 4) * 512 + frag_slot_b(kk, px & 15)] = (_Float16)x[e];
      }
    }
    __syncthreads();
    v16h a = lds_frag(&sA[w * 512 + lane * 16]);
#pragma unroll
    for (int p = 0; p < 4; ++p) {
      v16h bb = lds_frag(&sB[p * 512 + lane * 16]);
      acc[p] = wmma_f16(a, bb, acc[p]);
    }
  }
  // q-vs-v select is uniform per block (oc0 tile of 128 is entirely one side)
  _Float16* dst = ((oc0 < 256) ? q_out : v_out) + (size_t)b * HEADS * Ntok * DHEAD;
  const int chb = (oc0 & 255) + w * 16 + 8 * half;
  const int hstride = Ntok * DHEAD;
#pragma unroll
  for (int p = 0; p < 4; ++p) {
    int tok64 = (px0 + p * 16 + ln) * DHEAD;
#pragma unroll
    for (int r = 0; r < 8; ++r) {
      int ch = chb + r;
      int d = ch >> 2, h = ch & 3;
      dst[h * hstride + tok64 + d] = (_Float16)acc[p][r];
    }
  }
}

// ---------------------------------------------------------------------------
// Kernel 2: S[bh][n][m] = scale * sum_d q[bh][n][d]*mq[bh][m][d]
// Block: tile 64 n x 256 m, K = 64 (2 WMMA k-steps). All tiles frag-ready.
// ---------------------------------------------------------------------------
__global__ void attn_score_kernel(const _Float16* __restrict__ q,
                                  const _Float16* __restrict__ mq,
                                  float* __restrict__ S) {
  __shared__ alignas(32) _Float16 sQA[8 * 512];   // tile = kstep*4 + nsub
  __shared__ alignas(32) _Float16 sMQ[32 * 512];  // tile = kstep*16 + msub
  const int tid = threadIdx.x;
  const int w = tid >> 5, lane = tid & 31;
  const int half = lane >> 4, ln = lane & 15;
  const int n0 = blockIdx.x * 64;
  const int bh = blockIdx.y;

  const _Float16* qB = q + ((size_t)bh * NFEAT + n0) * DHEAD;
  const _Float16* mqB = mq + (size_t)bh * NMAP * DHEAD;
#pragma unroll
  for (int i = 0; i < 2; ++i) {  // q tile: 64 rows x 64 d, 8-wide runs
    int r = tid + i * 256;
    int row = r >> 3, d8 = (r & 7) * 8;
    v8h x = *(const v8h*)(qB + row * DHEAD + d8);
    *(v8h*)&sQA[((d8 >> 5) * 4 + (row >> 4)) * 512 + frag_slot_a(row & 15, d8 & 31)] = x;
  }
#pragma unroll
  for (int i = 0; i < 8; ++i) {  // mq: 256 m x 64 d, 8-wide runs
    int r = tid + i * 256;
    int m = r >> 3, d8 = (r & 7) * 8;
    v8h x = *(const v8h*)(mqB + m * DHEAD + d8);
    *(v8h*)&sMQ[((d8 >> 5) * 16 + (m >> 4)) * 512 + frag_slot_b(d8 & 31, m & 15)] = x;
  }
  __syncthreads();

  v8f acc[4][2] = {};
#pragma unroll
  for (int ks = 0; ks < 2; ++ks) {
    v16h bf[2];
#pragma unroll
    for (int mm = 0; mm < 2; ++mm)
      bf[mm] = lds_frag(&sMQ[(ks * 16 + 2 * w + mm) * 512 + lane * 16]);
#pragma unroll
    for (int ns = 0; ns < 4; ++ns) {
      v16h a = lds_frag(&sQA[(ks * 4 + ns) * 512 + lane * 16]);
#pragma unroll
      for (int mm = 0; mm < 2; ++mm)
        acc[ns][mm] = wmma_f16(a, bf[mm], acc[ns][mm]);
    }
  }
  float* Sb = S + (size_t)bh * NFEAT * NMAP;
#pragma unroll
  for (int ns = 0; ns < 4; ++ns)
#pragma unroll
    for (int mm = 0; mm < 2; ++mm)
#pragma unroll
      for (int r = 0; r < 8; ++r) {
        int n = n0 + ns * 16 + r + 8 * half;
        int m = (2 * w + mm) * 16 + ln;
        Sb[n * NMAP + m] = ATTN_SCALE * acc[ns][mm][r];
      }
}

// ---------------------------------------------------------------------------
// Kernel 3: row softmax over m (256) + feat_o = P1 * map_v.
// Block: 32 feat rows; 8 threads/row (each owns 32 contiguous cols) softmax via
// wave shuffles; WMMA over K=256. feat_o f16 as [b][n][c], c = d*4+h.
// ---------------------------------------------------------------------------
__global__ void row_softmax_feato_kernel(const float* __restrict__ S,
                                         const _Float16* __restrict__ mv,
                                         _Float16* __restrict__ feato) {
  __shared__ alignas(32) _Float16 sP[16 * 512];  // tile = kstep*2 + nsub
  __shared__ alignas(32) _Float16 sV[32 * 512];  // tile = kstep*4 + dsub
  const int tid = threadIdx.x;
  const int w = tid >> 5, lane = tid & 31;
  const int half = lane >> 4, ln = lane & 15;
  const int n0 = blockIdx.x * 32;
  const int bh = blockIdx.y;
  const int b = bh >> 2, h = bh & 3;

  const _Float16* mvB = mv + (size_t)bh * NMAP * DHEAD;
#pragma unroll
  for (int i = 0; i < 8; ++i) {  // map_v: 256 m x 64 d; 8-wide k(m)-runs
    int r = tid + i * 256;
    int d = r & 63, m8 = (r >> 6) * 8;
    v8h x;
#pragma unroll
    for (int j = 0; j < 8; ++j) x[j] = mvB[(m8 + j) * DHEAD + d];
    *(v8h*)&sV[((m8 >> 5) * 4 + (d >> 4)) * 512 + frag_slot_b(m8 & 31, d & 15)] = x;
  }

  const int row = tid >> 3, sub = tid & 7;  // 8 threads/row, 32 contiguous cols
  const float* srow = S + ((size_t)bh * NFEAT + n0 + row) * NMAP + sub * 32;
  float xv[32];
#pragma unroll
  for (int t = 0; t < 8; ++t) {
    v4f x = *(const v4f*)(srow + t * 4);
#pragma unroll
    for (int e = 0; e < 4; ++e) xv[t * 4 + e] = x[e];
  }
  float mx = xv[0];
#pragma unroll
  for (int i = 1; i < 32; ++i) mx = fmaxf(mx, xv[i]);
  for (int o = 1; o < 8; o <<= 1) mx = fmaxf(mx, __shfl_xor(mx, o, 32));
  float sm = 0.f;
#pragma unroll
  for (int i = 0; i < 32; ++i) sm += __expf(xv[i] - mx);
  for (int o = 1; o < 8; o <<= 1) sm += __shfl_xor(sm, o, 32);
  float inv = 1.0f / sm;
#pragma unroll
  for (int i8 = 0; i8 < 4; ++i8) {  // 4x 16B frag-ready stores
    int c = sub * 32 + i8 * 8;
    v8h p;
#pragma unroll
    for (int e = 0; e < 8; ++e) p[e] = (_Float16)(__expf(xv[i8 * 8 + e] - mx) * inv);
    *(v8h*)&sP[((c >> 5) * 2 + (row >> 4)) * 512 + frag_slot_a(row & 15, c & 31)] = p;
  }
  __syncthreads();

  const int ns = w >> 2, ds = w & 3;  // 2 n-subtiles x 4 d-subtiles = 8 waves
  v8f acc = {};
#pragma unroll
  for (int ks = 0; ks < 8; ++ks) {
    v16h a = lds_frag(&sP[(ks * 2 + ns) * 512 + lane * 16]);
    v16h bb = lds_frag(&sV[(ks * 4 + ds) * 512 + lane * 16]);
    acc = wmma_f16(a, bb, acc);
  }
  _Float16* fo = feato + (size_t)b * NFEAT * CIN;
#pragma unroll
  for (int r = 0; r < 8; ++r) {
    int n = n0 + ns * 16 + r + 8 * half;
    int d = ds * 16 + ln;
    fo[n * CIN + d * 4 + h] = (_Float16)acc[r];
  }
}

// ---------------------------------------------------------------------------
// Kernels 4a/4b: column softmax stats over j (9216 feat tokens), online max/sum
// with split-K partials. One thread per map token m.
// ---------------------------------------------------------------------------
__global__ void col_part_kernel(const float* __restrict__ S,
                                float* __restrict__ pmax,
                                float* __restrict__ psum) {
  const int bh = blockIdx.y, s = blockIdx.x, m = threadIdx.x;
  const int chunk = NFEAT / 16;
  float mx = -3.0e38f, sm = 0.f;
  const float* base = S + ((size_t)bh * NFEAT + s * chunk) * NMAP + m;
  for (int j = 0; j < chunk; ++j) {
    float x = base[j * NMAP];
    float nm = fmaxf(mx, x);
    sm = sm * __expf(mx - nm) + __expf(x - nm);
    mx = nm;
  }
  pmax[(bh * 16 + s) * NMAP + m] = mx;
  psum[(bh * 16 + s) * NMAP + m] = sm;
}

__global__ void col_reduce_kernel(const float* __restrict__ pmax,
                                  const float* __restrict__ psum,
                                  float* __restrict__ cmax,
                                  float* __restrict__ csum) {
  const int bh = blockIdx.x, m = threadIdx.x;
  float gm = -3.0e38f;
  for (int s = 0; s < 16; ++s)
    gm = fmaxf(gm, pmax[(bh * 16 + s) * NMAP + m]);
  float gs = 0.f;
  for (int s = 0; s < 16; ++s) {
    float pm = pmax[(bh * 16 + s) * NMAP + m];
    gs += psum[(bh * 16 + s) * NMAP + m] * __expf(pm - gm);
  }
  cmax[bh * NMAP + m] = gm;
  csum[bh * NMAP + m] = gs;
}

// ---------------------------------------------------------------------------
// Kernel 5: map_o[bh][i][d] = (sum_j exp(S[j][i]-cmax[i]) * fv[j][d]) / csum[i]
// 8 waves split the 9216-token K range; per-wave frag-ready staging of both
// exp(S)^T (A) and feat_v (B); cross-wave reduce via LDS ds_add_f32.
// map_o stored f16 as [b][i][c], c = d*4+h.
// ---------------------------------------------------------------------------
__global__ void map_o_kernel(const float* __restrict__ S,
                             const _Float16* __restrict__ fv,
                             const float* __restrict__ cmax,
                             const float* __restrict__ csum,
                             _Float16* __restrict__ mapo) {
  __shared__ alignas(32) _Float16 sE[8][512];       // per-wave A tile (16i x 32j)
  __shared__ alignas(32) _Float16 sFV[8][4 * 512];  // per-wave B tiles (32j x 64d)
  __shared__ float sAcc[16 * 64];
  const int tid = threadIdx.x;
  const int w = tid >> 5, lane = tid & 31;
  const int i0 = blockIdx.x * 16;
  const int bh = blockIdx.y;
  const int b = bh >> 2, h = bh & 3;

  for (int idx = tid; idx < 16 * 64; idx += 256) sAcc[idx] = 0.f;
  __syncthreads();

  const float* Sb = S + (size_t)bh * NFEAT * NMAP + i0;
  const float* cmB = cmax + bh * NMAP + i0;
  const _Float16* fvB = fv + (size_t)bh * NFEAT * DHEAD;
  v8f acc[4] = {};
  for (int j0 = w * 32; j0 < NFEAT; j0 += 256) {
    // A tile: exp(S[j][i] - colmax[i]) transposed into frag-ready layout
#pragma unroll
    for (int t = 0; t < 4; ++t) {
      int idx4 = lane + t * 32;
      int jj = idx4 >> 2, ii4 = (idx4 & 3) * 4;
      v4f x = *(const v4f*)(Sb + (j0 + jj) * NMAP + ii4);
      v4f cm = *(const v4f*)(cmB + ii4);
#pragma unroll
      for (int e = 0; e < 4; ++e)
        sE[w][frag_slot_a(ii4 + e, jj)] = (_Float16)__expf(x[e] - cm[e]);
    }
    // B tiles: 8-wide k(j)-runs -> single 16B frag-ready stores
#pragma unroll
    for (int t = 0; t < 8; ++t) {
      int r = lane + t * 32;
      int d = r & 63, j8 = (r >> 6) * 8;
      v8h x;
#pragma unroll
      for (int j = 0; j < 8; ++j) x[j] = fvB[(j0 + j8 + j) * DHEAD + d];
      *(v8h*)&sFV[w][(d >> 4) * 512 + frag_slot_b(j8, d & 15)] = x;
    }
    // same-wave LDS RAW: DS ops in-order per wave; no barrier needed
    v16h a = lds_frag(&sE[w][lane * 16]);
#pragma unroll
    for (int ds = 0; ds < 4; ++ds) {
      v16h bb = lds_frag(&sFV[w][ds * 512 + lane * 16]);
      acc[ds] = wmma_f16(a, bb, acc[ds]);
    }
  }
  const int half = lane >> 4, ln = lane & 15;
#pragma unroll
  for (int ds = 0; ds < 4; ++ds)
#pragma unroll
    for (int r = 0; r < 8; ++r)
      atomicAdd(&sAcc[(r + 8 * half) * 64 + ds * 16 + ln], acc[ds][r]);
  __syncthreads();
  _Float16* mo = mapo + (size_t)b * NMAP * CIN;
  for (int idx = tid; idx < 16 * 64; idx += 256) {
    int il = idx >> 6, d = idx & 63;
    int ig = i0 + il;
    float val = sAcc[idx] / csum[bh * NMAP + ig];
    mo[ig * CIN + d * 4 + h] = (_Float16)val;
  }
}

// ---------------------------------------------------------------------------
// Kernel 6: output projection. out[b][oc][n] = Wo[256,256] * Xo[b][n][256]
// Block: tile 64 oc x 64 n, K=256 in steps of 32; frag-ready LDS tiles,
// 8-wide k-runs for both W (f32->f16) and X (f16 passthrough) staging.
// ---------------------------------------------------------------------------
__global__ void out_proj_kernel(const _Float16* __restrict__ Xo,
                                const float* __restrict__ Wo,
                                float* __restrict__ out,
                                int Ntok) {
  __shared__ alignas(32) _Float16 sW[4 * 512];  // 4 A-tiles (oc-subtile)
  __shared__ alignas(32) _Float16 sX[4 * 512];  // 4 B-tiles (n-subtile)
  const int tid = threadIdx.x;
  const int w = tid >> 5, lane = tid & 31;
  const int half = lane >> 4, ln = lane & 15;
  const int n0 = blockIdx.x * 64;
  const int oc0 = blockIdx.y * 64;
  const int b = blockIdx.z;

  const int ocs = w >> 1, npair = w & 1;  // 4 oc-subtiles x (2x2 n-subtiles)
  v8f acc[2] = {};
  for (int k0 = 0; k0 < CIN; k0 += 32) {
    const float* WoB = Wo + oc0 * CIN + k0;
    const _Float16* XoB = Xo + ((size_t)b * Ntok + n0) * CIN + k0;
    __syncthreads();
    {  // W tile: 64 oc x 32 k, one 8-wide run per thread
      int oc = tid >> 2, k8 = (tid & 3) * 8;
      const float* src = WoB + oc * CIN + k8;
      *(v8h*)&sW[(oc >> 4) * 512 + frag_slot_a(oc & 15, k8)] =
          cvt_v8h(*(const v4f*)src, *(const v4f*)(src + 4));
    }
    {  // X tile: 64 n x 32 k, one 8-wide run per thread
      int n = tid >> 2, k8 = (tid & 3) * 8;
      v8h x = *(const v8h*)(XoB + n * CIN + k8);
      *(v8h*)&sX[(n >> 4) * 512 + frag_slot_b(k8, n & 15)] = x;
    }
    __syncthreads();
    v16h a = lds_frag(&sW[ocs * 512 + lane * 16]);
#pragma unroll
    for (int nn = 0; nn < 2; ++nn) {
      v16h bb = lds_frag(&sX[(npair * 2 + nn) * 512 + lane * 16]);
      acc[nn] = wmma_f16(a, bb, acc[nn]);
    }
  }
  float* ob = out + (size_t)b * CIN * Ntok;
#pragma unroll
  for (int nn = 0; nn < 2; ++nn)
#pragma unroll
    for (int r = 0; r < 8; ++r) {
      int oc = oc0 + ocs * 16 + r + 8 * half;
      int n = n0 + (npair * 2 + nn) * 16 + ln;
      ob[oc * Ntok + n] = acc[nn][r];
    }
}

// ---------------------------------------------------------------------------
extern "C" void kernel_launch(void* const* d_in, const int* in_sizes, int n_in,
                              void* d_out, int out_size, void* d_ws, size_t ws_size,
                              hipStream_t stream) {
  (void)in_sizes; (void)n_in; (void)out_size; (void)ws_size;
  const float* feat = (const float*)d_in[0];
  const float* smap = (const float*)d_in[1];
  const float* Wfqv = (const float*)d_in[2];
  const float* Wmqv = (const float*)d_in[3];
  const float* Wfo  = (const float*)d_in[4];
  const float* Wmo  = (const float*)d_in[5];
  float* out = (float*)d_out;

  uint8_t* ws = (uint8_t*)d_ws;
  size_t off = 0;
  auto take = [&](size_t bytes) -> void* {
    void* p = ws + off;
    off += (bytes + 255) & ~(size_t)255;
    return p;
  };
  _Float16* featq = (_Float16*)take((size_t)NBH * NFEAT * DHEAD * 2);
  _Float16* featv = (_Float16*)take((size_t)NBH * NFEAT * DHEAD * 2);
  _Float16* mapq  = (_Float16*)take((size_t)NBH * NMAP * DHEAD * 2);
  _Float16* mapv  = (_Float16*)take((size_t)NBH * NMAP * DHEAD * 2);
  float*    Smat  = (float*)take((size_t)NBH * NFEAT * NMAP * 4);
  _Float16* feato = (_Float16*)take((size_t)NBATCH * NFEAT * CIN * 2);
  _Float16* mapo  = (_Float16*)take((size_t)NBATCH * NMAP * CIN * 2);
  float* pmax = (float*)take((size_t)NBH * 16 * NMAP * 4);
  float* psum = (float*)take((size_t)NBH * 16 * NMAP * 4);
  float* cmax = (float*)take((size_t)NBH * NMAP * 4);
  float* csum = (float*)take((size_t)NBH * NMAP * 4);

  dim3 blk(256);
  qv_proj_kernel<<<dim3(NFEAT / 64, 4, NBATCH), blk, 0, stream>>>(feat, Wfqv, featq, featv, NFEAT);
  qv_proj_kernel<<<dim3(NMAP / 64, 4, NBATCH), blk, 0, stream>>>(smap, Wmqv, mapq, mapv, NMAP);
  attn_score_kernel<<<dim3(NFEAT / 64, NBH), blk, 0, stream>>>(featq, mapq, Smat);
  row_softmax_feato_kernel<<<dim3(NFEAT / 32, NBH), blk, 0, stream>>>(Smat, mapv, feato);
  col_part_kernel<<<dim3(16, NBH), blk, 0, stream>>>(Smat, pmax, psum);
  col_reduce_kernel<<<dim3(NBH), blk, 0, stream>>>(pmax, psum, cmax, csum);
  map_o_kernel<<<dim3(NMAP / 16, NBH), blk, 0, stream>>>(Smat, featv, cmax, csum, mapo);
  out_proj_kernel<<<dim3(NFEAT / 64, 4, NBATCH), blk, 0, stream>>>(feato, Wfo, out, NFEAT);
  out_proj_kernel<<<dim3(NMAP / 64, 4, NBATCH), blk, 0, stream>>>(
      mapo, Wmo, out + (size_t)NBATCH * CIN * NFEAT, NMAP);
}
